// MultiHeadAttention_13168369729738
// MI455X (gfx1250) — matrix operand order repaired and checked by reading, not compiled, not run on hardware
//
#include <hip/hip_runtime.h>

#define B_ 2
#define S_ 2048
#define E_ 1024
#define H_ 16
#define D_ 64

typedef __attribute__((ext_vector_type(16))) __bf16        v16bf;
typedef __attribute__((ext_vector_type(4)))  __bf16        v4bf;
typedef __attribute__((ext_vector_type(8)))  float         v8f;
typedef __attribute__((ext_vector_type(4)))  float         v4f;
typedef __attribute__((ext_vector_type(4)))  unsigned int  v4u;

union Frag {           // 32 bytes: one bf16 WMMA A/B operand per lane
    v16bf v;
    __bf16 h[16];
    v4bf  h4[4];
    unsigned short u[16];
    v4u q[2];
};

__device__ __forceinline__ unsigned short f2bf(float f) {
    union { __bf16 b; unsigned short u; } cv;
    cv.b = (__bf16)f;          // native packed RNE convert on gfx1250
    return cv.u;
}

__device__ __forceinline__ v8f wmma_bf16(v16bf a, v16bf b, v8f c) {
    return __builtin_amdgcn_wmma_f32_16x16x32_bf16(
        false, a, false, b, (short)0, c, false, false);
}

__device__ __forceinline__ v8f v8f_zero() {
    v8f z;
#pragma unroll
    for (int i = 0; i < 8; ++i) z[i] = 0.0f;
    return z;
}

// async global -> LDS copy, 16 bytes (tracked by ASYNCcnt)
__device__ __forceinline__ void async_b128(const unsigned short* lds,
                                           const unsigned short* gsrc) {
    asm volatile("global_load_async_to_lds_b128 %0, %1, off"
                 :: "v"((unsigned)(size_t)lds),
                    "v"((unsigned long long)(size_t)gsrc)
                 : "memory");
}

// reductions within each 16-lane half of a wave32 (xor masks < 16 never cross halves)
__device__ __forceinline__ float rowmax16(float v) {
#pragma unroll
    for (int m = 8; m >= 1; m >>= 1) v = fmaxf(v, __shfl_xor(v, m, 32));
    return v;
}
__device__ __forceinline__ float rowsum16(float v) {
#pragma unroll
    for (int m = 8; m >= 1; m >>= 1) v += __shfl_xor(v, m, 32);
    return v;
}

// ---------------------------------------------------------------- weights f32 -> bf16 (with scale fold)
__global__ void cvt_f32_bf16(const float* __restrict__ src,
                             unsigned short* __restrict__ dst, int n, float scale) {
    int i = blockIdx.x * blockDim.x + threadIdx.x;
    int stride = gridDim.x * blockDim.x;
    for (; i < n; i += stride) dst[i] = f2bf(src[i] * scale);
}

// ---------------------------------------------------------------- QKV projection
// One wave handles 16 tokens of one (b,h): C(16x64) = X(16x64) @ W^T for Q,K,V.
// 1/sqrt(d) is folded into Wq at conversion time. V stored transposed: Vt[b][h][d][s].
__global__ __launch_bounds__(128) void project_qkv(
    const float* __restrict__ q, const float* __restrict__ k, const float* __restrict__ v,
    const unsigned short* __restrict__ Wqb, const unsigned short* __restrict__ Wkb,
    const unsigned short* __restrict__ Wvb,
    unsigned short* __restrict__ Qp, unsigned short* __restrict__ Kp,
    unsigned short* __restrict__ Vt)
{
    const int lane = threadIdx.x, wy = threadIdx.y;
    const int tile = blockIdx.x * 4 + wy;          // 4096 tiles
    const int S16 = S_ / 16;
    const int b  = tile / (H_ * S16);
    const int rem = tile % (H_ * S16);
    const int h  = rem / S16;
    const int s0 = (rem % S16) * 16;
    const int half = lane >> 4, lr = lane & 15;

    const float* xs[3] = {q, k, v};
    const unsigned short* ws3[3] = {Wqb, Wkb, Wvb};

#pragma unroll
    for (int mat = 0; mat < 3; ++mat) {
        const float* x = xs[mat];
        Frag a[2];
#pragma unroll
        for (int ks = 0; ks < 2; ++ks) {
            const float* base = x + ((size_t)(b * S_ + s0 + lr)) * E_ + h * D_
                                  + ks * 32 + half * 8;
            a[ks].h4[0] = __builtin_convertvector(*(const v4f*)(base),      v4bf);
            a[ks].h4[1] = __builtin_convertvector(*(const v4f*)(base + 4),  v4bf);
            a[ks].h4[2] = __builtin_convertvector(*(const v4f*)(base + 16), v4bf);
            a[ks].h4[3] = __builtin_convertvector(*(const v4f*)(base + 20), v4bf);
        }
        const unsigned short* W = ws3[mat];
#pragma unroll
        for (int nt = 0; nt < 4; ++nt) {
            v8f acc = v8f_zero();
#pragma unroll
            for (int ks = 0; ks < 2; ++ks) {
                Frag bb;   // B[k][n] = W[n][k]: row-major W -> contiguous per-lane
                const unsigned short* wp = W + (size_t)(nt * 16 + lr) * D_
                                             + ks * 32 + half * 8;
                bb.q[0] = *(const v4u*)(wp);
                bb.q[1] = *(const v4u*)(wp + 16);
                acc = wmma_bf16(a[ks].v, bb.v, acc);
            }
#pragma unroll
            for (int r = 0; r < 8; ++r) {
                const int M = r + half * 8;
                const unsigned short val = f2bf(acc[r]);
                if (mat == 2) {
                    Vt[((size_t)(b * H_ + h) * D_ + nt * 16 + lr) * S_ + s0 + M] = val;
                } else {
                    unsigned short* dst = (mat == 0) ? Qp : Kp;
                    dst[((size_t)(b * H_ + h) * S_ + s0 + M) * D_ + nt * 16 + lr] = val;
                }
            }
        }
    }
}

// ---------------------------------------------------------------- flash attention
// 4 waves/block share one (b,h); K/V chunks staged to LDS once per block via
// async global->LDS loads (ASYNCcnt), double-buffered; each wave owns a
// 16-query tile and walks Sk in chunks of 32.
__global__ __launch_bounds__(128) void attention(
    const unsigned short* __restrict__ Qp, const unsigned short* __restrict__ Kp,
    const unsigned short* __restrict__ Vt, unsigned short* __restrict__ Oint)
{
    __shared__ __align__(16) unsigned short Kbuf[2][32][72]; // 144B row stride (bank spread)
    __shared__ __align__(16) unsigned short Vbuf[2][64][40]; // 80B row stride
    __shared__ __align__(16) unsigned short plds[4][16][32]; // per-wave P staging

    const int lane = threadIdx.x, wy = threadIdx.y;
    const int tid  = wy * 32 + lane;               // 0..127
    const int tile = blockIdx.x * 4 + wy;
    const int S16 = S_ / 16;
    const int b  = tile / (H_ * S16);
    const int rem = tile % (H_ * S16);
    const int h  = rem / S16;
    const int s0 = (rem % S16) * 16;
    const int half = lane >> 4, lr = lane & 15;
    const size_t bh = (size_t)(b * H_ + h);        // identical for all 4 waves

    const unsigned short* qrow  = Qp + (bh * S_ + s0 + lr) * D_;
    const unsigned short* kbase = Kp + bh * S_ * (size_t)D_;
    const unsigned short* vbase = Vt + bh * D_ * (size_t)S_;

    // loader mapping: each thread copies 32B of K and 32B of V per chunk
    const int krow = tid >> 2, kcol = (tid & 3) * 16;   // K chunk: 32 x 64
    const int vrow = tid >> 1, vcol = (tid & 1) * 16;   // V chunk: 64 x 32

    Frag qa[2];
#pragma unroll
    for (int ks = 0; ks < 2; ++ks) {
        const unsigned short* p = qrow + ks * 32 + half * 8;
        qa[ks].q[0] = *(const v4u*)(p);
        qa[ks].q[1] = *(const v4u*)(p + 16);
    }

    v8f o[4];
    float mrow[8], lrow[8];
#pragma unroll
    for (int t = 0; t < 4; ++t) o[t] = v8f_zero();
#pragma unroll
    for (int r = 0; r < 8; ++r) { mrow[r] = -3.0e38f; lrow[r] = 0.0f; }

    // prologue: stage chunk 0 into buffer 0 (4 async b128 per thread)
    {
        const unsigned short* ksrc = kbase + (size_t)krow * D_ + kcol;
        async_b128(&Kbuf[0][krow][kcol],     ksrc);
        async_b128(&Kbuf[0][krow][kcol + 8], ksrc + 8);
        const unsigned short* vsrc = vbase + (size_t)vrow * S_ + vcol;
        async_b128(&Vbuf[0][vrow][vcol],     vsrc);
        async_b128(&Vbuf[0][vrow][vcol + 8], vsrc + 8);
    }

    for (int j0 = 0, it = 0; j0 < S_; j0 += 32, ++it) {
        const int cur = it & 1;
        __syncthreads();                       // prev readers of buf cur^1 are done
        if (j0 + 32 < S_) {                    // stage next chunk into the other buffer
            const int nb = cur ^ 1;
            const unsigned short* ksrc = kbase + (size_t)(j0 + 32 + krow) * D_ + kcol;
            async_b128(&Kbuf[nb][krow][kcol],     ksrc);
            async_b128(&Kbuf[nb][krow][kcol + 8], ksrc + 8);
            const unsigned short* vsrc = vbase + (size_t)vrow * S_ + j0 + 32 + vcol;
            async_b128(&Vbuf[nb][vrow][vcol],     vsrc);
            async_b128(&Vbuf[nb][vrow][vcol + 8], vsrc + 8);
            asm volatile("s_wait_asynccnt 0x4" ::: "memory");  // cur chunk landed (in-order)
        } else {
            asm volatile("s_wait_asynccnt 0x0" ::: "memory");
        }
        __syncthreads();                       // cur buffer visible to all waves

        // scores S(16x32) = Qs @ K^T  (K fragments from LDS)
        v8f sf[2];
#pragma unroll
        for (int nt = 0; nt < 2; ++nt) {
            v8f acc = v8f_zero();
#pragma unroll
            for (int ks = 0; ks < 2; ++ks) {
                Frag bb;
                const unsigned short* p = &Kbuf[cur][nt * 16 + lr][ks * 32 + half * 8];
                bb.q[0] = *(const v4u*)(p);
                bb.q[1] = *(const v4u*)(p + 16);
                acc = wmma_bf16(qa[ks].v, bb.v, acc);
            }
            sf[nt] = acc;
        }
        // online softmax (rows live in 16-lane halves; same mapping as O accum)
#pragma unroll
        for (int r = 0; r < 8; ++r) {
            float cm = fmaxf(sf[0][r], sf[1][r]);
            cm = rowmax16(cm);
            const float mn = fmaxf(mrow[r], cm);
            const float sc = __expf(mrow[r] - mn);
            const float p0 = __expf(sf[0][r] - mn);
            const float p1 = __expf(sf[1][r] - mn);
            const float rs = rowsum16(p0 + p1);
            lrow[r] = lrow[r] * sc + rs;
            mrow[r] = mn;
#pragma unroll
            for (int t = 0; t < 4; ++t) o[t][r] = o[t][r] * sc;
            plds[wy][r + half * 8][lr]      = f2bf(p0);
            plds[wy][r + half * 8][16 + lr] = f2bf(p1);
        }
        asm volatile("s_wait_dscnt 0" ::: "memory");
        Frag pf;   // reload P as A fragment
        pf.q[0] = *(const v4u*)&plds[wy][lr][half * 8];
        pf.q[1] = *(const v4u*)&plds[wy][lr][half * 8 + 16];
        // O += P @ V   (V fragments from LDS)
#pragma unroll
        for (int t = 0; t < 4; ++t) {
            Frag bb;
            const unsigned short* p = &Vbuf[cur][t * 16 + lr][half * 8];
            bb.q[0] = *(const v4u*)(p);
            bb.q[1] = *(const v4u*)(p + 16);
            o[t] = wmma_bf16(pf.v, bb.v, o[t]);
        }
    }
#pragma unroll
    for (int r = 0; r < 8; ++r) {
        const float inv = 1.0f / lrow[r];
        const int M = r + half * 8;
#pragma unroll
        for (int t = 0; t < 4; ++t) {
            Oint[((size_t)b * S_ + s0 + M) * E_ + h * D_ + t * 16 + lr]
                = f2bf(o[t][r] * inv);
        }
    }
}

// ---------------------------------------------------------------- output projection
// Y(4096x1024) = O(4096x1024) @ Wo^T + bo ; wave tile = 16x64, K walked by 32.
__global__ __launch_bounds__(128) void out_proj(
    const unsigned short* __restrict__ Oint, const unsigned short* __restrict__ Wob,
    const float* __restrict__ bo, float* __restrict__ out)
{
    const int lane = threadIdx.x, wy = threadIdx.y;
    const int tile = blockIdx.x * 4 + wy;          // 256 * 16 tiles
    const int m0 = (tile >> 4) * 16;
    const int n0 = (tile & 15) * 64;
    const int half = lane >> 4, lr = lane & 15;

    v8f acc[4];
#pragma unroll
    for (int t = 0; t < 4; ++t) acc[t] = v8f_zero();

    const unsigned short* arow = Oint + (size_t)(m0 + lr) * E_;
    for (int k0 = 0; k0 < E_; k0 += 32) {
        Frag a;
        const unsigned short* p = arow + k0 + half * 8;
        a.q[0] = *(const v4u*)(p);
        a.q[1] = *(const v4u*)(p + 16);
#pragma unroll
        for (int nt = 0; nt < 4; ++nt) {
            Frag bb;   // B[k][n] = Wo[n][k]
            const unsigned short* wp = Wob + (size_t)(n0 + nt * 16 + lr) * E_
                                           + k0 + half * 8;
            bb.q[0] = *(const v4u*)(wp);
            bb.q[1] = *(const v4u*)(wp + 16);
            acc[nt] = wmma_bf16(a.v, bb.v, acc[nt]);
        }
    }
#pragma unroll
    for (int nt = 0; nt < 4; ++nt) {
        const float bias = bo[n0 + nt * 16 + lr];
#pragma unroll
        for (int r = 0; r < 8; ++r) {
            const int M = r + half * 8;
            out[(size_t)(m0 + M) * E_ + n0 + nt * 16 + lr] = acc[nt][r] + bias;
        }
    }
}

// ---------------------------------------------------------------- launcher
extern "C" void kernel_launch(void* const* d_in, const int* in_sizes, int n_in,
                              void* d_out, int out_size, void* d_ws, size_t ws_size,
                              hipStream_t stream) {
    const float* key_  = (const float*)d_in[0];
    const float* query = (const float*)d_in[1];
    const float* value = (const float*)d_in[2];
    const float* Wq    = (const float*)d_in[3];
    const float* Wk    = (const float*)d_in[4];
    const float* Wv    = (const float*)d_in[5];
    const float* Wo    = (const float*)d_in[6];
    const float* bo    = (const float*)d_in[7];
    float* out = (float*)d_out;

    // bf16 workspace layout
    unsigned short* ws = (unsigned short*)d_ws;
    const size_t NP = (size_t)B_ * H_ * S_ * D_;   // 4,194,304
    unsigned short* Qp   = ws;
    unsigned short* Kp   = Qp + NP;
    unsigned short* Vt   = Kp + NP;
    unsigned short* Oint = Vt + NP;
    unsigned short* Wqb  = Oint + (size_t)B_ * S_ * E_;
    unsigned short* Wkb  = Wqb + D_ * D_;
    unsigned short* Wvb  = Wkb + D_ * D_;
    unsigned short* Wob  = Wvb + D_ * D_;

    cvt_f32_bf16<<<16, 256, 0, stream>>>(Wq, Wqb, D_ * D_, 0.125f);  // fold 1/sqrt(64)
    cvt_f32_bf16<<<16, 256, 0, stream>>>(Wk, Wkb, D_ * D_, 1.0f);
    cvt_f32_bf16<<<16, 256, 0, stream>>>(Wv, Wvb, D_ * D_, 1.0f);
    cvt_f32_bf16<<<1024, 256, 0, stream>>>(Wo, Wob, E_ * E_, 1.0f);

    dim3 blk(32, 4);
    project_qkv<<<1024, blk, 0, stream>>>(query, key_, value, Wqb, Wkb, Wvb, Qp, Kp, Vt);
    attention<<<1024, blk, 0, stream>>>(Qp, Kp, Vt, Oint);
    out_proj<<<1024, blk, 0, stream>>>(Oint, Wob, bo, out);
}
